// SynthesizerAttention_52029233824466
// MI455X (gfx1250) — compile-verified, matrix operand order checked by reading, not run on hardware
//
#include <hip/hip_runtime.h>
#include <hip/hip_bf16.h>
#include <stdint.h>

// ---------------------------------------------------------------------------
// SynthesizerAttention on MI455X (gfx1250), wave32 + WMMA bf16.
//   B=2, T=2048, C=1024, nh=16, hs=64
// Strategy: one-time bf16 conversion of x and (transposed) weights into ws
// (all L2-resident on 192MB L2), then LDS-free register-blocked WMMA GEMMs
// loading fragments directly via global_load_b128, plus a fused flash-style
// causal attention that never materializes the (B,nh,T,T) score tensor.
// ---------------------------------------------------------------------------

typedef __bf16 bf16_t;
typedef __attribute__((ext_vector_type(16))) __bf16 v16bf;
typedef __attribute__((ext_vector_type(8)))  __bf16 v8bf;
typedef __attribute__((ext_vector_type(8)))  float  v8f;
typedef __attribute__((ext_vector_type(4)))  float  v4f;

#define NB   2
#define NT   2048
#define NC   1024
#define NH   16
#define HS   64
#define NM   (NB * NT)          // 4096 rows in the dense GEMMs
#define NEGV (-1e10f)
#define LDSTR 40                // padded LDS row stride (80B, 16B aligned)

__device__ __forceinline__ bf16_t f2bf(float f) {
    union { float f; uint32_t u; } c; c.f = f;
    uint32_t r = c.u + 0x7fffu + ((c.u >> 16) & 1u);   // round-to-nearest-even
    union { uint16_t u; bf16_t b; } o; o.u = (uint16_t)(r >> 16);
    return o.b;
}

// Load a 16x32 bf16 operand tile in WMMA A/B register layout (works for
// global or LDS pointers). base points at (row0, k0) of a row-major tile
// with `stride` elements per row. Per ISA layout: lane L holds row (L&15);
// K elements (h*8..h*8+7) and (16+h*8..23+h*8) where h = L>>4
// -> two 16-byte loads per lane, concatenated with shufflevector.
__device__ __forceinline__ v16bf frag_ld(const bf16_t* base, int stride) {
    const int lane = threadIdx.x & 31;
    const bf16_t* p = base + (size_t)(lane & 15) * stride + ((lane >> 4) << 3);
    v8bf lo = *(const v8bf*)(p);
    v8bf hi = *(const v8bf*)(p + 16);
    return __builtin_shufflevector(lo, hi, 0, 1, 2, 3, 4, 5, 6, 7,
                                           8, 9, 10, 11, 12, 13, 14, 15);
}

__device__ __forceinline__ v8f wmma_bf16(v16bf a, v16bf b, v8f c) {
    return __builtin_amdgcn_wmma_f32_16x16x32_bf16(
        /*neg_a=*/false, a, /*neg_b=*/false, b,
        /*c_mod=*/(short)0, c, /*reuse_a=*/false, /*reuse_b=*/false);
}

// Cross-lane XOR reduction step via ds_swizzle (group-of-32 bit mode:
// offset[14:10]=xor_mask, offset[4:0]=and_mask). IMM must be an ICE.
template <int IMM>
__device__ __forceinline__ float swz_xor(float v) {
    union { float f; int i; } c; c.f = v;
    c.i = __builtin_amdgcn_ds_swizzle(c.i, IMM);
    return c.f;
}
#define SWZ1 0x041F   // xor 1, and 0x1f
#define SWZ2 0x081F   // xor 2
#define SWZ4 0x101F   // xor 4
#define SWZ8 0x201F   // xor 8

__device__ __forceinline__ float half_max(float v) {   // max over 16-lane half
    v = fmaxf(v, swz_xor<SWZ1>(v));
    v = fmaxf(v, swz_xor<SWZ2>(v));
    v = fmaxf(v, swz_xor<SWZ4>(v));
    v = fmaxf(v, swz_xor<SWZ8>(v));
    return v;
}
__device__ __forceinline__ float half_sum(float v) {   // sum over 16-lane half
    v += swz_xor<SWZ1>(v);
    v += swz_xor<SWZ2>(v);
    v += swz_xor<SWZ4>(v);
    v += swz_xor<SWZ8>(v);
    return v;
}

// ---------------------------------------------------------------------------
// Prep kernels: one-time fp32 -> bf16 conversions into workspace.
// ---------------------------------------------------------------------------
__global__ void conv_x_kernel(const float* __restrict__ src,
                              bf16_t* __restrict__ dst, int n) {
    int i = (blockIdx.x * 256 + threadIdx.x) * 8;
    if (i < n) {
        v4f a = *(const v4f*)(src + i);
        v4f b = *(const v4f*)(src + i + 4);
        v8bf o;
#pragma unroll
        for (int j = 0; j < 4; ++j) { o[j] = f2bf(a[j]); o[4 + j] = f2bf(b[j]); }
        *(v8bf*)(dst + i) = o;
    }
}

// W (K x N = 1024x1024 row-major) -> Wt (N x K, bf16). Reads coalesced in n.
__global__ void conv_wt_kernel(const float* __restrict__ W,
                               bf16_t* __restrict__ Wt) {
    int idx = blockIdx.x * 256 + threadIdx.x;    // over (K/4)*N = 262144
    int n  = idx & (NC - 1);
    int k  = (idx >> 10) << 2;
#pragma unroll
    for (int j = 0; j < 4; ++j)
        Wt[(size_t)n * NC + k + j] = f2bf(W[(size_t)(k + j) * NC + n]);
}

// w2 (hs x T fp32) -> w2T (T x hs bf16): score-GEMM B-fragments become
// contiguous 16B loads (lane = key column, elements = hs).
__global__ void prep_w2_kernel(const float* __restrict__ w2,
                               bf16_t* __restrict__ w2t) {
    int i = blockIdx.x * 256 + threadIdx.x;      // over T*HS
    if (i < NT * HS) {
        int t = i >> 6, d = i & 63;
        w2t[i] = f2bf(w2[(size_t)d * NT + t]);
    }
}

// ---------------------------------------------------------------------------
// LDS-free dense GEMM: Out[M x N] = A[M x K](bf16) @ Wt[N x K](bf16)^T + bias
//   OUT_BF16: emit bf16 ws tensor vs fp32 final output
//   FUSE_Q:   epilogue relu(acc + bias + b1[n&63])  (synthesizer query path)
// Block = 256 threads (8 waves), block tile 128(M) x 128(N).
// Wave (wm 0..3, wn 0..1) owns 32(M) x 64(N) = 2x4 WMMA tiles.
// All fragments are direct global_load_b128 pairs; x/W bf16 copies are
// L2-resident (8MB + 2MB inside 192MB L2), so no LDS staging is needed.
// ---------------------------------------------------------------------------
template <bool OUT_BF16, bool FUSE_Q>
__global__ __launch_bounds__(256)
void gemm_bf16_kernel(const bf16_t* __restrict__ A,
                      const bf16_t* __restrict__ Wt,
                      const float* __restrict__ bias,
                      const float* __restrict__ b1,
                      void* __restrict__ Out, int M, int N, int K) {
    const int tid  = threadIdx.x;
    const int lane = tid & 31;
    const int wave = tid >> 5;
    const int wm   = wave & 3;                 // 4 waves across M
    const int wn   = wave >> 2;                // 2 waves across N
    const int m0   = blockIdx.y * 128 + wm * 32;
    const int n0   = blockIdx.x * 128 + wn * 64;

    v8f acc[2][4];
#pragma unroll
    for (int i = 0; i < 2; ++i)
#pragma unroll
        for (int j = 0; j < 4; ++j)
#pragma unroll
            for (int e = 0; e < 8; ++e) acc[i][j][e] = 0.0f;

    const bf16_t* Abase = A  + (size_t)m0 * K;
    const bf16_t* Wbase = Wt + (size_t)n0 * K;

    for (int k0 = 0; k0 < K; k0 += 64) {
        if (k0 + 64 < K) {                     // prefetch next K-slab
            __builtin_prefetch(Abase + (size_t)(lane & 15) * K + k0 + 64, 0, 1);
            __builtin_prefetch(Wbase + (size_t)(lane & 15) * K + k0 + 64, 0, 1);
        }
#pragma unroll
        for (int u = 0; u < 2; ++u) {          // K unrolled x2 (32 each)
            const int kk = k0 + u * 32;
            v16bf a0 = frag_ld(Abase + kk,          K);
            v16bf a1 = frag_ld(Abase + 16 * K + kk, K);
            v16bf b0 = frag_ld(Wbase + 0  * K + kk, K);
            v16bf b1 = frag_ld(Wbase + 16 * K + kk, K);
            v16bf b2 = frag_ld(Wbase + 32 * K + kk, K);
            v16bf b3 = frag_ld(Wbase + 48 * K + kk, K);
            acc[0][0] = wmma_bf16(a0, b0, acc[0][0]);
            acc[0][1] = wmma_bf16(a0, b1, acc[0][1]);
            acc[0][2] = wmma_bf16(a0, b2, acc[0][2]);
            acc[0][3] = wmma_bf16(a0, b3, acc[0][3]);
            acc[1][0] = wmma_bf16(a1, b0, acc[1][0]);
            acc[1][1] = wmma_bf16(a1, b1, acc[1][1]);
            acc[1][2] = wmma_bf16(a1, b2, acc[1][2]);
            acc[1][3] = wmma_bf16(a1, b3, acc[1][3]);
        }
    }

    // Epilogue (C layout: VGPR r -> row r + 8*(lane>>4); lane&15 -> column).
    const int half = lane >> 4;
    const int colx = lane & 15;
#pragma unroll
    for (int tn = 0; tn < 4; ++tn) {
        int n = n0 + tn * 16 + colx;
        float bs = bias[n];
        if constexpr (FUSE_Q) bs += b1[n & 63];
#pragma unroll
        for (int tm = 0; tm < 2; ++tm)
#pragma unroll
            for (int r = 0; r < 8; ++r) {
                int m = m0 + tm * 16 + r + 8 * half;
                float v = acc[tm][tn][r] + bs;
                if constexpr (FUSE_Q) v = fmaxf(v, 0.0f);
                if constexpr (OUT_BF16)
                    ((bf16_t*)Out)[(size_t)m * N + n] = f2bf(v);
                else
                    ((float*)Out)[(size_t)m * N + n] = v;
            }
    }
}

// ---------------------------------------------------------------------------
// Fused causal synthesizer attention (flash style, never materializes att).
// Grid: (T/128, B*nh). Block = 8 waves; wave w owns query rows q0..q0+15.
// Per 32-key block: scores = Q(16x64) @ w2T-block (4 chained WMMAs, K=64),
// +b2, causal mask (only on the single diagonal block per wave), online
// softmax (ds_swizzle row reductions over lane halves), P re-laid-out via
// wave-private LDS into A-fragment form, then y += P(16x32) @ V(32x64)
// against an LDS-staged V^T tile (4 WMMAs). All branch conditions are
// wave-uniform, so EXEC stays all-ones around WMMA / swizzle per the ISA.
// ---------------------------------------------------------------------------
__global__ __launch_bounds__(256)
void attn_kernel(const bf16_t* __restrict__ Q, const bf16_t* __restrict__ V,
                 const bf16_t* __restrict__ W2T, const float* __restrict__ b2,
                 bf16_t* __restrict__ Y) {
    __shared__ __align__(16) bf16_t vt[HS * LDSTR];            // V^T [d][key]
    __shared__ __align__(16) bf16_t plds[8 * 16 * LDSTR];      // per-wave P

    const int tid  = threadIdx.x;
    const int lane = tid & 31;
    const int wave = tid >> 5;
    const int half = lane >> 4;
    const int lcol = lane & 15;

    const int bh   = blockIdx.y;        // 0..B*NH-1
    const int b    = bh >> 4;
    const int head = bh & 15;
    const int q0   = blockIdx.x * 128 + wave * 16;

    // Q fragments for this wave's 16 rows (K = hs = 64 -> two 16x32 frags)
    const bf16_t* qbase = Q + ((size_t)(b * NT + q0)) * NC + head * HS;
    v16bf qf0 = frag_ld(qbase,      NC);
    v16bf qf1 = frag_ld(qbase + 32, NC);

    float mi[8], li[8];
    v8f   yacc[4];
#pragma unroll
    for (int r = 0; r < 8; ++r) { mi[r] = -3.0e38f; li[r] = 0.0f; }
#pragma unroll
    for (int n = 0; n < 4; ++n)
#pragma unroll
        for (int e = 0; e < 8; ++e) yacc[n][e] = 0.0f;

    bf16_t* pw = &plds[wave * 16 * LDSTR];
    // Staging indices: one v8bf global load + 8 transposed ds_store_b16.
    const int skey = tid >> 3;              // 0..31
    const int sdg  = (tid & 7) * 8;         // d group of 8
    const bf16_t* vsrc = V + ((size_t)(b * NT)) * NC + head * HS + sdg;

    const int blockKend = blockIdx.x * 128 + 128;   // block max causal key + 1

    for (int j = 0; j < blockKend; j += 32) {
        __syncthreads();
        // ---- cooperative stage of V^T tile: [d in 0..63][key in 0..31] ----
        {
            v8bf vv = *(const v8bf*)(vsrc + (size_t)(j + skey) * NC);
#pragma unroll
            for (int u = 0; u < 8; ++u)
                vt[(sdg + u) * LDSTR + skey] = vv[u];
        }
        __syncthreads();

        if (j < q0 + 16) {                          // wave-uniform causal skip
            // ---- scores: two 16x16 tiles, K=64 via chained WMMA ----
            v8f s[2];
#pragma unroll
            for (int t = 0; t < 2; ++t) {
                const bf16_t* wb = W2T + (size_t)(j + t * 16) * HS;
                v16bf w0 = frag_ld(wb,      HS);
                v16bf w1 = frag_ld(wb + 32, HS);
                v8f z;
#pragma unroll
                for (int e = 0; e < 8; ++e) z[e] = 0.0f;
                z = wmma_bf16(qf0, w0, z);
                z = wmma_bf16(qf1, w1, z);
                s[t] = z;
            }
            // ---- + b2 ----
#pragma unroll
            for (int t = 0; t < 2; ++t) {
                float bb = b2[j + t * 16 + lcol];
#pragma unroll
                for (int r = 0; r < 8; ++r) s[t][r] += bb;
            }
            // ---- causal mask: only the diagonal block can clip ----
            if (j + 31 > q0) {                      // wave-uniform
#pragma unroll
                for (int t = 0; t < 2; ++t) {
                    int key = j + t * 16 + lcol;
#pragma unroll
                    for (int r = 0; r < 8; ++r) {
                        int row = q0 + r + 8 * half;
                        if (key > row) s[t][r] = NEGV;
                    }
                }
            }
            // ---- online softmax; row r+8*half lives on one 16-lane half ---
            float p0[8], p1[8], alpha[8];
#pragma unroll
            for (int r = 0; r < 8; ++r) {
                float rm   = half_max(fmaxf(s[0][r], s[1][r]));
                float mnew = fmaxf(mi[r], rm);
                float al   = __expf(mi[r] - mnew);
                float e0   = __expf(s[0][r] - mnew);
                float e1   = __expf(s[1][r] - mnew);
                float rs   = half_sum(e0 + e1);
                li[r] = li[r] * al + rs;
                mi[r] = mnew;
                alpha[r] = al; p0[r] = e0; p1[r] = e1;
            }
#pragma unroll
            for (int n = 0; n < 4; ++n)
#pragma unroll
                for (int r = 0; r < 8; ++r) yacc[n][r] *= alpha[r];

            // ---- P: C-layout -> A-layout via wave-private LDS ----
#pragma unroll
            for (int r = 0; r < 8; ++r) {
                pw[(r + 8 * half) * LDSTR + lcol]      = f2bf(p0[r]);
                pw[(r + 8 * half) * LDSTR + 16 + lcol] = f2bf(p1[r]);
            }
            v16bf pa = frag_ld(pw, LDSTR);          // 16x32 A fragment

            // ---- y += P @ V  (4 N-tiles over hs) ----
#pragma unroll
            for (int n = 0; n < 4; ++n) {
                v16bf vb = frag_ld(&vt[(n * 16) * LDSTR], LDSTR);
                yacc[n] = wmma_bf16(pa, vb, yacc[n]);
            }
        }
    }

    // ---- normalize and store y (bf16 ws tensor, (B*T, C) layout) ----
#pragma unroll
    for (int n = 0; n < 4; ++n)
#pragma unroll
        for (int r = 0; r < 8; ++r) {
            int row = q0 + r + 8 * half;
            int col = head * HS + n * 16 + lcol;
            Y[((size_t)(b * NT + row)) * NC + col] = f2bf(yacc[n][r] / li[r]);
        }
}

// ---------------------------------------------------------------------------
// Host launcher. Inputs: x, W1, bw1, b1, w2, b2, Wv, bv, Wp, bp (all fp32).
// ---------------------------------------------------------------------------
extern "C" void kernel_launch(void* const* d_in, const int* in_sizes, int n_in,
                              void* d_out, int out_size, void* d_ws, size_t ws_size,
                              hipStream_t stream) {
    const float* x   = (const float*)d_in[0];
    const float* W1  = (const float*)d_in[1];
    const float* bw1 = (const float*)d_in[2];
    const float* b1  = (const float*)d_in[3];
    const float* w2  = (const float*)d_in[4];
    const float* b2  = (const float*)d_in[5];
    const float* Wv  = (const float*)d_in[6];
    const float* bv  = (const float*)d_in[7];
    const float* Wp  = (const float*)d_in[8];
    const float* bp  = (const float*)d_in[9];
    float* out = (float*)d_out;

    // Workspace layout (bf16 tensors)
    char* ws = (char*)d_ws;
    const size_t tens  = (size_t)NM * NC * sizeof(bf16_t);  // 8 MB
    const size_t wtens = (size_t)NC * NC * sizeof(bf16_t);  // 2 MB
    bf16_t* xb   = (bf16_t*)(ws);
    bf16_t* q_ws = (bf16_t*)(ws + tens);
    bf16_t* v_ws = (bf16_t*)(ws + 2 * tens);
    bf16_t* y_ws = (bf16_t*)(ws + 3 * tens);
    bf16_t* W1t  = (bf16_t*)(ws + 4 * tens);
    bf16_t* Wvt  = (bf16_t*)(ws + 4 * tens + wtens);
    bf16_t* Wpt  = (bf16_t*)(ws + 4 * tens + 2 * wtens);
    bf16_t* w2t  = (bf16_t*)(ws + 4 * tens + 3 * wtens);

    // 0) one-time bf16 conversions (all destinations L2-resident)
    conv_x_kernel<<<(NM * NC / 8 + 255) / 256, 256, 0, stream>>>(x, xb, NM * NC);
    conv_wt_kernel<<<NC * NC / 4 / 256, 256, 0, stream>>>(W1, W1t);
    conv_wt_kernel<<<NC * NC / 4 / 256, 256, 0, stream>>>(Wv, Wvt);
    conv_wt_kernel<<<NC * NC / 4 / 256, 256, 0, stream>>>(Wp, Wpt);
    prep_w2_kernel<<<(NT * HS + 255) / 256, 256, 0, stream>>>(w2, w2t);

    dim3 ggrid(NC / 128, NM / 128);     // (8, 32)
    // 1) q = relu(x @ W1 + bw1 + b1) -> bf16
    gemm_bf16_kernel<true, true><<<ggrid, 256, 0, stream>>>(
        xb, W1t, bw1, b1, q_ws, NM, NC, NC);
    // 2) v = x @ Wv + bv -> bf16
    gemm_bf16_kernel<true, false><<<ggrid, 256, 0, stream>>>(
        xb, Wvt, bv, nullptr, v_ws, NM, NC, NC);
    // 3) fused causal attention -> y (bf16)
    dim3 agrid(NT / 128, NB * NH);      // (16, 32)
    attn_kernel<<<agrid, 256, 0, stream>>>(q_ws, v_ws, w2t, b2, y_ws);
    // 4) out = y @ Wp + bp -> fp32
    gemm_bf16_kernel<false, false><<<ggrid, 256, 0, stream>>>(
        y_ws, Wpt, bp, nullptr, out, NM, NC, NC);
}
